// GraphConvolution_45552423141532
// MI455X (gfx1250) — matrix operand order repaired and checked by reading, not compiled, not run on hardware
//
#include <hip/hip_runtime.h>
#include <hip/hip_bf16.h>

// ---------------------------------------------------------------------------
// GCN layer: out = segment_sum(edge_val * (X@W)[col], row) + bias
//   N = 100000 nodes, E = 3.2M edges, D_IN = D_OUT = 256
// Phase 1: WMMA bf16 GEMM  support = X @ W   (bf16 result, L2-resident 51 MB)
// Phase 2: per-edge wave32 scatter with global_atomic_add_f32
// ---------------------------------------------------------------------------

#define D 256
#define TILE_K 32

typedef __attribute__((ext_vector_type(16))) __bf16 v16bf;
typedef __attribute__((ext_vector_type(8)))  __bf16 v8bf;
typedef __attribute__((ext_vector_type(8)))  float  v8f;

// --- Kernel 1: W[k][n] fp32 -> Wt[n][k] bf16 (pre-transpose so B fragments
// are contiguous in K per lane) -----------------------------------------------
__global__ __launch_bounds__(256) void gcn_wt_bf16(const float* __restrict__ W,
                                                   __bf16* __restrict__ Wt) {
  int idx = blockIdx.x * 256 + threadIdx.x;   // 65536 elements
  int k = idx >> 8;
  int n = idx & (D - 1);
  Wt[(size_t)n * D + k] = (__bf16)W[idx];
}

// --- Kernel 2: support = X @ W via v_wmma_f32_16x16x32_bf16 ------------------
// Block = 256 threads (8 waves). Block computes a 16-row x 256-col tile.
// Wave w owns columns [w*32, w*32+32) as two 16x16 WMMA subtiles.
__global__ __launch_bounds__(256) void gcn_gemm_wmma(
    const float* __restrict__ X, const __bf16* __restrict__ Wt,
    __bf16* __restrict__ support) {
  __shared__ __align__(16) __bf16 Abf[16][TILE_K];   // 1 KB A staging tile

  const int tid   = threadIdx.x;
  const int wave  = tid >> 5;
  const int lane  = tid & 31;
  const int nlane = lane & 15;
  const int half  = lane >> 4;
  const int rowBase = blockIdx.x * 16;   // 6250 * 16 == 100000 exactly

  const int n0 = wave * 32;
  const int n1 = n0 + 16;

  v8f acc0 = {};
  v8f acc1 = {};

#pragma unroll
  for (int k0 = 0; k0 < D; k0 += TILE_K) {
    // Cooperative stage: 16x32 fp32 -> bf16 in LDS (512 elems, 2 per thread)
    {
      const int idx = tid * 2;
      const int r   = idx >> 5;     // 0..15
      const int kk  = idx & 31;     // 0..31, even
      const float2 xv =
          *(const float2*)(X + (size_t)(rowBase + r) * D + k0 + kk);
      Abf[r][kk]     = (__bf16)xv.x;
      Abf[r][kk + 1] = (__bf16)xv.y;
    }
    __syncthreads();

    // A fragment (16-bit A 16x32): lane m = lane%16,
    // element e <-> K = half*16 + e (matches B)
    const v8bf alo = *reinterpret_cast<const v8bf*>(&Abf[nlane][half * 16]);
    const v8bf ahi = *reinterpret_cast<const v8bf*>(&Abf[nlane][half * 16 + 8]);
    const v16bf afrag = __builtin_shufflevector(
        alo, ahi, 0, 1, 2, 3, 4, 5, 6, 7, 8, 9, 10, 11, 12, 13, 14, 15);

    // B fragments (32x16): lane n = lane%16, element e <-> K = half*16 + e.
    // Wt is [N][K] bf16 so the 16 K-values are contiguous per lane.
    const __bf16* b0p = Wt + (size_t)(n0 + nlane) * D + k0 + half * 16;
    const v8bf b0lo = *(const v8bf*)(b0p);
    const v8bf b0hi = *(const v8bf*)(b0p + 8);
    const v16bf bfrag0 = __builtin_shufflevector(
        b0lo, b0hi, 0, 1, 2, 3, 4, 5, 6, 7, 8, 9, 10, 11, 12, 13, 14, 15);

    const __bf16* b1p = Wt + (size_t)(n1 + nlane) * D + k0 + half * 16;
    const v8bf b1lo = *(const v8bf*)(b1p);
    const v8bf b1hi = *(const v8bf*)(b1p + 8);
    const v16bf bfrag1 = __builtin_shufflevector(
        b1lo, b1hi, 0, 1, 2, 3, 4, 5, 6, 7, 8, 9, 10, 11, 12, 13, 14, 15);

    acc0 = __builtin_amdgcn_wmma_f32_16x16x32_bf16(
        false, afrag, false, bfrag0, (short)0, acc0, false, false);
    acc1 = __builtin_amdgcn_wmma_f32_16x16x32_bf16(
        false, afrag, false, bfrag1, (short)0, acc1, false, false);

    __syncthreads();
  }

  // C/D layout: VGPR v, lanes 0-15 -> M=v, lanes 16-31 -> M=v+8, N=lane%16
#pragma unroll
  for (int v = 0; v < 8; ++v) {
    const int m = rowBase + half * 8 + v;
    support[(size_t)m * D + n0 + nlane] = (__bf16)acc0[v];
    support[(size_t)m * D + n1 + nlane] = (__bf16)acc1[v];
  }
}

// --- Kernel 3: out[i][f] = bias[f] (d_out is poisoned by the harness) --------
__global__ __launch_bounds__(256) void gcn_bias_init(
    const float* __restrict__ bias, float* __restrict__ out, int total) {
  int i = blockIdx.x * 256 + threadIdx.x;
  if (i < total) out[i] = bias[i & (D - 1)];
}

// --- Kernel 4: per-edge scatter: out[row] += val * support[col] --------------
// One wave32 per edge; each lane handles 8 contiguous features (16B bf16
// gather, all-L2-resident) and issues 8 native f32 atomic adds.
__global__ __launch_bounds__(256) void gcn_spmm_scatter(
    const int* __restrict__ erow, const int* __restrict__ ecol,
    const float* __restrict__ eval, const __bf16* __restrict__ support,
    float* __restrict__ out, int nEdges) {
  const int gwave = (int)((blockIdx.x * 256 + threadIdx.x) >> 5);
  const int lane  = threadIdx.x & 31;
  if (gwave >= nEdges) return;

  const int   r = erow[gwave];   // wave-uniform -> scalarized loads
  const int   c = ecol[gwave];
  const float v = eval[gwave];

  const v8bf s = *(const v8bf*)(support + (size_t)c * D + lane * 8);
  float* op = out + (size_t)r * D + lane * 8;
#pragma unroll
  for (int i = 0; i < 8; ++i) {
    unsafeAtomicAdd(op + i, v * (float)s[i]);   // global_atomic_add_f32
  }
}

// ---------------------------------------------------------------------------
extern "C" void kernel_launch(void* const* d_in, const int* in_sizes, int n_in,
                              void* d_out, int out_size, void* d_ws,
                              size_t ws_size, hipStream_t stream) {
  const float* X    = (const float*)d_in[0];  // [100000, 256] f32
  const int*   erow = (const int*)d_in[1];    // [E] i32
  const int*   ecol = (const int*)d_in[2];    // [E] i32
  const float* eval = (const float*)d_in[3];  // [E] f32
  const float* W    = (const float*)d_in[4];  // [256, 256] f32
  const float* bias = (const float*)d_in[5];  // [256] f32
  float* out = (float*)d_out;                 // [100000, 256] f32

  const int nEdges = in_sizes[1];
  const int nNodes = in_sizes[0] / D;
  const int total  = nNodes * D;

  char* ws = (char*)d_ws;
  __bf16* Wt      = (__bf16*)ws;                  // 256*256*2   = 128 KB
  __bf16* support = (__bf16*)(ws + (D * D * 2));  // 100000*256*2 = 51.2 MB

  // 1) W -> Wt (bf16, transposed)
  gcn_wt_bf16<<<(D * D) / 256, 256, 0, stream>>>(W, Wt);

  // 2) support = X @ W via WMMA (6250 blocks of 16 rows)
  gcn_gemm_wmma<<<nNodes / 16, 256, 0, stream>>>(X, Wt, support);

  // 3) out = bias (broadcast)
  gcn_bias_init<<<(total + 255) / 256, 256, 0, stream>>>(bias, out, total);

  // 4) edge scatter (one wave per edge -> 8 waves per 256-thread block)
  const int nBlocks = (nEdges + 7) / 8;
  gcn_spmm_scatter<<<nBlocks, 256, 0, stream>>>(erow, ecol, eval, support, out,
                                                nEdges);
}
